// ARDecoder_18227841204400
// MI455X (gfx1250) — matrix operand order repaired and checked by reading, not compiled, not run on hardware
//
#include <hip/hip_runtime.h>
#include <hip/hip_bf16.h>

// ---------------------------------------------------------------------------
// ARDecoder (MHA encoder summary + GRU autoregressive decoder) for gfx1250.
// GEMMs: v_wmma_f32_16x16x32_bf16 fed by tensor_load_to_lds (TDM) B-tiles.
// Attention: two-pass flash style; V pre-transposed so every WMMA operand
// loads K-contiguous (b128 clauses); P tile transposed through LDS for P@V.
// ---------------------------------------------------------------------------

typedef __attribute__((ext_vector_type(16))) __bf16 v16bf;
typedef __attribute__((ext_vector_type(8)))  float  v8f;
typedef __attribute__((ext_vector_type(4)))  unsigned v4u;
typedef __attribute__((ext_vector_type(8)))  int    v8i;
typedef __attribute__((ext_vector_type(4)))  int    v4i;

#define DEV __device__ __forceinline__

constexpr int B  = 64;
constexpr int L  = 512;
constexpr int C  = 512;
constexpr int H  = 4;
constexpr int D  = 128;       // C / H
constexpr int S  = 128;
constexpr int T  = 64;        // future_steps (python scalar in setup_inputs)
constexpr int TC = 3 * C;     // qkv row width = 1536
constexpr int XD = C + S;     // GRU input width = 640
constexpr int G3 = 3 * C;     // gate width = 1536
constexpr float SCALE = 0.08838834764831845f;  // 1/sqrt(D)

#if __has_builtin(__builtin_amdgcn_tensor_load_to_lds) && \
    __has_builtin(__builtin_amdgcn_s_wait_tensorcnt)
#define HAS_TDM 1
#else
#define HAS_TDM 0
#endif

// ---------------- WMMA fragment helpers (contiguous k = 16*(lane/16) + s) -----------

// A fragment: 16x32 bf16, A row-major, leading dim lda.
// lanes 0-15: M=lane, K=0..15; lanes 16-31: M=lane-16, K=16..31.
DEV v16bf load_a_frag(const __bf16* A, int lda) {
  const int lane = threadIdx.x & 31;
  const int m  = lane & 15;
  const int kb = (lane < 16) ? 0 : 16;
  v16bf a;
#pragma unroll
  for (int v = 0; v < 8; ++v) {
    const int k = kb + 2 * v;
    a[2 * v]     = A[(size_t)m * lda + k];
    a[2 * v + 1] = A[(size_t)m * lda + k + 1];
  }
  return a;
}

// B fragment: 32x16 where B[k][n] = W[n][k] (W row-major [N,K], ldw).
// lanes 0-15 hold K=0..15, lanes 16-31 hold K=16..31; N=lane&15. 16 contiguous
// bf16 per lane -> 2x b128 loads (global or ds depending on address space).
DEV v16bf load_bT_frag(const __bf16* W, int ldw) {
  const int lane = threadIdx.x & 31;
  const int n  = lane & 15;
  const int kb = (lane < 16) ? 0 : 16;
  v16bf b;
#pragma unroll
  for (int v = 0; v < 8; ++v) {
    b[2 * v]     = W[(size_t)n * ldw + kb + 2 * v];
    b[2 * v + 1] = W[(size_t)n * ldw + kb + 2 * v + 1];
  }
  return b;
}

DEV v8f wmma_bf16(v16bf a, v16bf b, v8f c) {
  return __builtin_amdgcn_wmma_f32_16x16x32_bf16(
      /*neg_a=*/false, a, /*neg_b=*/false, b,
      /*c_mod=*/(short)0, c, /*reuse_a=*/false, /*reuse_b=*/false);
}

// reductions over the 16 lanes that share a C/D-tile row
DEV float half_max(float v) {
  v = fmaxf(v, __shfl_xor(v, 1));
  v = fmaxf(v, __shfl_xor(v, 2));
  v = fmaxf(v, __shfl_xor(v, 4));
  v = fmaxf(v, __shfl_xor(v, 8));
  return v;
}
DEV float half_sum(float v) {
  v += __shfl_xor(v, 1);
  v += __shfl_xor(v, 2);
  v += __shfl_xor(v, 4);
  v += __shfl_xor(v, 8);
  return v;
}

// ---------------- Tensor Data Mover: 64(N) x 32(K) bf16 tile -> LDS ---------
// D# descriptor per CDNA5 ISA 8.3/8.4: 2D tensor, data_size=2B,
// tensor_dim0 = tile_dim0 = 32 (contiguous K), tensor_dim1 = tile_dim1 = 64
// (N rows), tensor_dim0_stride = ldw elements. Rows land densely in LDS:
// Bs[64][32]. Wave-level op (EXEC ignored), issued by wave 0 only.
DEV void tdm_load_b_tile(const __bf16* gsrc, unsigned ldw_elems, unsigned lds_off) {
#if HAS_TDM
  const unsigned long long ga = (unsigned long long)(uintptr_t)gsrc;
  v4u g0;
  g0[0] = 1u;                                   // count=1, user mode
  g0[1] = lds_off;                              // lds_addr (bytes)
  g0[2] = (unsigned)ga;                         // global_addr[31:0]
  g0[3] = (unsigned)((ga >> 32) & 0x01FFFFFFu)  // global_addr[56:32]
          | (2u << 30);                         // type = 2 ("image")
  v8i g1;
  g1[0] = (int)(1u << 16);        // workgroup_mask=0, data_size=1 (2 bytes)
  g1[1] = (int)(32u << 16);       // tensor_dim0[15:0]=32 at bits[31:16]
  g1[2] = (int)(64u << 16);       // tensor_dim0 hi=0 | tensor_dim1[15:0]=64
  g1[3] = (int)(32u << 16);       // tensor_dim1 hi=0 | tile_dim0=32
  g1[4] = (int)64;                // tile_dim1=64 | tile_dim2=0
  g1[5] = (int)ldw_elems;         // tensor_dim0_stride[31:0]
  g1[6] = 0;                      // stride hi | tensor_dim1_stride lo
  g1[7] = 0;
  v4i z4 = {0, 0, 0, 0};
#if __clang_major__ >= 23
  v8i z8 = {0, 0, 0, 0, 0, 0, 0, 0};
  __builtin_amdgcn_tensor_load_to_lds(g0, g1, z4, z4, z8, 0);
#else
  __builtin_amdgcn_tensor_load_to_lds(g0, g1, z4, z4, 0);
#endif
#else
  (void)gsrc; (void)ldw_elems; (void)lds_off;
#endif
}

// ---------------- small utility kernels ------------------------------------

__global__ void k_cvt_f32_bf16(const float* __restrict__ s, __bf16* __restrict__ d, int n) {
  int i = blockIdx.x * blockDim.x + threadIdx.x;
  if (i < n) d[i] = (__bf16)s[i];
}

__global__ void k_zero_f32(float* __restrict__ d, int n) {
  int i = blockIdx.x * blockDim.x + threadIdx.x;
  if (i < n) d[i] = 0.0f;
}

// V slice of qkv [b, l, 2C + h*D + d]  ->  vT [((b*H + h)*D + d) * L + l]
// so the P@V B-operand becomes K(=l)-contiguous.
__global__ void k_transpose_v(const __bf16* __restrict__ qkv,
                              __bf16* __restrict__ vT) {
  int i = blockIdx.x * blockDim.x + threadIdx.x;   // over B*L*C
  if (i >= B * L * C) return;
  const int hd = i % C;            // h*D + d
  const int l  = (i / C) % L;
  const int b  = i / (L * C);
  vT[((size_t)b * C + hd) * L + l] = qkv[((size_t)b * L + l) * TC + 2 * C + hd];
}

// ---------------- TDM-fed WMMA GEMM:  C = A[M,K] @ W[N,K]^T + bias ----------
// Macro-tile: (MW*16) rows of M  x  64 cols of N, one wave per 16-row strip.
// B tile (64x32 bf16, 4KB) is TDM-DMA'd to LDS, double buffered; all MW waves
// share it. A fragments stream from global (b128 clauses, +prefetch).
// grid = (N/64, M/(MW*16)); block = MW*32.

template <int MW, int WRITE_F32, int WRITE_BF16>
__global__ void k_gemm_tdm(const __bf16* __restrict__ A, int lda,
                           const __bf16* __restrict__ W, int ldw,
                           const float* __restrict__ bias,
                           float* __restrict__ Cf, __bf16* __restrict__ Cb,
                           int ldc, int K) {
  __shared__ __bf16 Bs[2][64 * 32];

  const int n0 = blockIdx.x * 64;
  const int m0 = blockIdx.y * (MW * 16);
  const int w  = threadIdx.x >> 5;   // wave id
  const int lane = threadIdx.x & 31;
  const __bf16* Arow = A + (size_t)(m0 + w * 16) * lda;
  const __bf16* Wrow = W + (size_t)n0 * ldw;

  v8f acc[4];
#pragma unroll
  for (int j = 0; j < 4; ++j) acc[j] = (v8f){};

#if HAS_TDM
  if (w == 0)
    tdm_load_b_tile(Wrow, (unsigned)ldw, (unsigned)(uintptr_t)&Bs[0][0]);
  int buf = 0;
  for (int k = 0; k < K; k += 32, buf ^= 1) {
    if (w == 0) __builtin_amdgcn_s_wait_tensorcnt(0);
    __syncthreads();                       // B tile visible to all waves
    if (w == 0 && k + 32 < K)
      tdm_load_b_tile(Wrow + k + 32, (unsigned)ldw,
                      (unsigned)(uintptr_t)&Bs[buf ^ 1][0]);
    __builtin_prefetch(Arow + k + 128, 0, 0);   // global_prefetch_b8
    v16bf a = load_a_frag(Arow + k, lda);
#pragma unroll
    for (int j = 0; j < 4; ++j) {
      v16bf b = load_bT_frag(&Bs[buf][j * 16 * 32], 32);
      acc[j] = wmma_bf16(a, b, acc[j]);
    }
    __syncthreads();                       // compute done before buf reuse
  }
#else
  // no-TDM fallback: cooperative synchronous staging of the B tile
  int buf = 0;
  for (int k = 0; k < K; k += 32, buf ^= 1) {
    for (int idx = threadIdx.x; idx < 64 * 32; idx += MW * 32) {
      const int nn = idx >> 5, kk = idx & 31;
      Bs[buf][idx] = Wrow[(size_t)nn * ldw + k + kk];
    }
    __syncthreads();
    v16bf a = load_a_frag(Arow + k, lda);
#pragma unroll
    for (int j = 0; j < 4; ++j) {
      v16bf b = load_bT_frag(&Bs[buf][j * 16 * 32], 32);
      acc[j] = wmma_bf16(a, b, acc[j]);
    }
    __syncthreads();
  }
#endif

  const int n  = lane & 15;
  const int mb = (lane < 16) ? 0 : 8;
#pragma unroll
  for (int j = 0; j < 4; ++j) {
    const float bv = bias ? bias[n0 + j * 16 + n] : 0.0f;
#pragma unroll
    for (int r = 0; r < 8; ++r) {
      float v = acc[j][r] + bv;
      const size_t off = (size_t)(m0 + w * 16 + mb + r) * ldc + n0 + j * 16 + n;
      if (WRITE_F32)  Cf[off] = v;
      if (WRITE_BF16) Cb[off] = (__bf16)v;
    }
  }
}

// ---------------- attention -------------------------------------------------

// 16x16 score tile (already scaled): S = Q[q0:q0+16, :D] . K[k-tile]^T
DEV v8f score_tile(const __bf16* qb, const __bf16* kb) {
  v8f s = {};
#pragma unroll
  for (int d = 0; d < D; d += 32) {
    v16bf a = load_a_frag(qb + d, TC);
    v16bf b = load_bT_frag(kb + d, TC);
    s = wmma_bf16(a, b, s);
  }
#pragma unroll
  for (int r = 0; r < 8; ++r) s[r] *= SCALE;
  return s;
}

// pass 1: per (b,h,q) online row-max and row-sumexp. grid (L/16, H, B), block 32.
__global__ void k_attn_pass1(const __bf16* __restrict__ qkv,
                             float* __restrict__ rowmax, float* __restrict__ rowsum) {
  const int q0 = blockIdx.x * 16, h = blockIdx.y, b = blockIdx.z;
  const int lane = threadIdx.x;
  const __bf16* qb = qkv + ((size_t)b * L + q0) * TC + h * D;

  float m8[8], s8[8];
#pragma unroll
  for (int r = 0; r < 8; ++r) { m8[r] = -3.0e38f; s8[r] = 0.0f; }

  for (int k0 = 0; k0 < L; k0 += 16) {
    const __bf16* kb = qkv + ((size_t)b * L + k0) * TC + C + h * D;
    v8f sc = score_tile(qb, kb);
#pragma unroll
    for (int r = 0; r < 8; ++r) {
      float tmax = half_max(sc[r]);
      float nm   = fmaxf(m8[r], tmax);
      float tsum = half_sum(__expf(sc[r] - nm));
      s8[r] = s8[r] * __expf(m8[r] - nm) + tsum;
      m8[r] = nm;
    }
  }
  if ((lane & 15) == 0) {
    const int half = lane >> 4;
#pragma unroll
    for (int r = 0; r < 8; ++r) {
      const size_t idx = ((size_t)(b * H + h)) * L + q0 + half * 8 + r;
      rowmax[idx] = m8[r];
      rowsum[idx] = s8[r];
    }
  }
}

// pass 2: exact P tiles -> O = P@V (WMMA, P transposed via LDS, V from vT so
// the B operand is K-contiguous) + colsum atomics. grid (L/16, H, B), block 32.
__global__ void k_attn_pass2(const __bf16* __restrict__ qkv,
                             const __bf16* __restrict__ vT,
                             const float* __restrict__ rowmax,
                             const float* __restrict__ rowsum,
                             float* __restrict__ colsum,
                             __bf16* __restrict__ attnO) {
  __shared__ __bf16 Pt[16][40];  // 16x32 P tile, padded rows

  const int q0 = blockIdx.x * 16, h = blockIdx.y, b = blockIdx.z;
  const int lane = threadIdx.x;
  const int half = lane >> 4, n = lane & 15;
  const __bf16* qb  = qkv + ((size_t)b * L + q0) * TC + h * D;
  const __bf16* vTb = vT + ((size_t)b * C + h * D) * L;   // [D rows][L cols]

  float rm[8], ri[8];
#pragma unroll
  for (int r = 0; r < 8; ++r) {
    const size_t idx = ((size_t)(b * H + h)) * L + q0 + half * 8 + r;
    rm[r] = rowmax[idx];
    ri[r] = 1.0f / rowsum[idx];
  }

  v8f o[8];
#pragma unroll
  for (int t = 0; t < 8; ++t) o[t] = (v8f){};

  for (int kk = 0; kk < L; kk += 32) {
#pragma unroll
    for (int co = 0; co < 32; co += 16) {
      const __bf16* kb = qkv + ((size_t)b * L + kk + co) * TC + C + h * D;
      v8f sc = score_tile(qb, kb);
      float cs = 0.0f;
#pragma unroll
      for (int r = 0; r < 8; ++r) {
        float p = __expf(sc[r] - rm[r]) * ri[r];
        Pt[half * 8 + r][co + n] = (__bf16)p;
        cs += p;
      }
      cs += __shfl_xor(cs, 16);            // combine both row halves of column
      if (lane < 16)
        atomicAdd(&colsum[(size_t)b * L + kk + co + lane], cs);
    }
    // P (16x32) as WMMA A fragment, straight out of LDS (single wave: DS ops
    // from one wave are in-order, no barrier needed)
    v16bf pa = load_a_frag(&Pt[0][0], 40);
#pragma unroll
    for (int nt = 0; nt < 8; ++nt) {
      // B[k][n] = V[kk+k][nt*16+n] = vT[(nt*16+n)*L + kk + k]  (K-contiguous)
      v16bf bv = load_bT_frag(vTb + (size_t)(nt * 16) * L + kk, L);
      o[nt] = wmma_bf16(pa, bv, o[nt]);
    }
  }

#pragma unroll
  for (int nt = 0; nt < 8; ++nt)
#pragma unroll
    for (int r = 0; r < 8; ++r)
      attnO[((size_t)b * L + q0 + half * 8 + r) * C + h * D + nt * 16 + n] =
          (__bf16)o[nt][r];
}

// ---------------- summary / softmax over keys -------------------------------
__global__ void k_summary(const float* __restrict__ colsum,
                          const float* __restrict__ attnC,
                          const float* __restrict__ context,
                          const float* __restrict__ static_embed,
                          __bf16* __restrict__ summ_bf,
                          __bf16* __restrict__ xbuf) {
  __shared__ float buf[512];
  __shared__ float red[256];
  __shared__ float mxsh, ssh;
  const int b = blockIdx.x, tid = threadIdx.x;

  float c0 = colsum[(size_t)b * L + tid]       * (1.0f / (H * L));
  float c1 = colsum[(size_t)b * L + 256 + tid] * (1.0f / (H * L));
  buf[tid] = c0; buf[tid + 256] = c1;
  red[tid] = fmaxf(c0, c1);
  __syncthreads();
  for (int st = 128; st > 0; st >>= 1) {
    if (tid < st) red[tid] = fmaxf(red[tid], red[tid + st]);
    __syncthreads();
  }
  if (tid == 0) mxsh = red[0];
  __syncthreads();
  float e0 = __expf(c0 - mxsh), e1 = __expf(c1 - mxsh);
  buf[tid] = e0; buf[tid + 256] = e1;
  red[tid] = e0 + e1;
  __syncthreads();
  for (int st = 128; st > 0; st >>= 1) {
    if (tid < st) red[tid] += red[tid + st];
    __syncthreads();
  }
  if (tid == 0) ssh = red[0];
  __syncthreads();
  const float inv = 1.0f / ssh;

  for (int c = tid; c < C; c += 256) {
    float acc = 0.0f, mean = 0.0f;
    for (int k = 0; k < L; ++k) {
      acc  += buf[k] * attnC[((size_t)b * L + k) * C + c];
      mean += context[((size_t)b * L + k) * C + c];
    }
    summ_bf[(size_t)b * C + c] = (__bf16)(acc * inv);
    xbuf[(size_t)b * XD + c]   = (__bf16)(mean * (1.0f / L));
  }
  for (int s = tid; s < S; s += 256)
    xbuf[(size_t)b * XD + C + s] = (__bf16)static_embed[(size_t)b * S + s];
}

// ---------------- GRU pointwise + head --------------------------------------

__global__ void k_gru_update(const float* __restrict__ gi, const float* __restrict__ gh,
                             float* __restrict__ h, __bf16* __restrict__ hbf) {
  const int idx = blockIdx.x * blockDim.x + threadIdx.x;
  if (idx >= B * C) return;
  const int b = idx / C, c = idx % C;
  const float* gib = gi + (size_t)b * G3;
  const float* ghb = gh + (size_t)b * G3;
  const float i_r = gib[c], i_z = gib[C + c], i_n = gib[2 * C + c];
  const float h_r = ghb[c], h_z = ghb[C + c], h_n = ghb[2 * C + c];
  const float r = 1.0f / (1.0f + __expf(-(i_r + h_r)));
  const float z = 1.0f / (1.0f + __expf(-(i_z + h_z)));
  const float nn = tanhf(i_n + r * h_n);
  const float hv = (1.0f - z) * nn + z * h[idx];
  h[idx]   = hv;
  hbf[idx] = (__bf16)hv;
}

__global__ void k_head_step(const float* __restrict__ h,
                            const float* __restrict__ W_head,
                            const float* __restrict__ b_head,
                            const float* __restrict__ W_tf,
                            const float* __restrict__ b_tf,
                            __bf16* __restrict__ xbuf,
                            float* __restrict__ out, int t) {
  __shared__ float red[256];
  __shared__ float predsh;
  const int b = blockIdx.x, tid = threadIdx.x;
  float p = h[(size_t)b * C + tid] * W_head[tid] +
            h[(size_t)b * C + 256 + tid] * W_head[256 + tid];
  red[tid] = p;
  __syncthreads();
  for (int st = 128; st > 0; st >>= 1) {
    if (tid < st) red[tid] += red[tid + st];
    __syncthreads();
  }
  if (tid == 0) {
    const float pr = red[0] + b_head[0];
    predsh = pr;
    out[(size_t)b * T + t] = pr;   // output layout [B, T, 1]
  }
  __syncthreads();
  const float pr = predsh;
  for (int c = tid; c < C; c += 256) {
    const float v = fmaxf(0.0f, pr * W_tf[c] + b_tf[c]);
    xbuf[(size_t)b * XD + c] = (__bf16)v;
  }
}

// ---------------- host driver ------------------------------------------------

static inline size_t ws_alloc(size_t& off, size_t bytes) {
  size_t r = off;
  off += (bytes + 255) & ~(size_t)255;
  return r;
}

extern "C" void kernel_launch(void* const* d_in, const int* in_sizes, int n_in,
                              void* d_out, int out_size, void* d_ws, size_t ws_size,
                              hipStream_t stream) {
  (void)in_sizes; (void)n_in; (void)out_size; (void)ws_size;
  const float* context = (const float*)d_in[0];
  const float* statice = (const float*)d_in[1];
  const float* W_in    = (const float*)d_in[2];
  const float* b_in    = (const float*)d_in[3];
  const float* W_out   = (const float*)d_in[4];
  const float* b_out   = (const float*)d_in[5];
  const float* W_init  = (const float*)d_in[6];
  const float* b_init  = (const float*)d_in[7];
  const float* W_ih    = (const float*)d_in[8];
  const float* b_ih    = (const float*)d_in[9];
  const float* W_hh    = (const float*)d_in[10];
  const float* b_hh    = (const float*)d_in[11];
  const float* W_tf    = (const float*)d_in[12];
  const float* b_tf    = (const float*)d_in[13];
  const float* W_head  = (const float*)d_in[14];
  const float* b_head  = (const float*)d_in[15];
  float* out = (float*)d_out;

  char* ws = (char*)d_ws;
  size_t off = 0;
  __bf16* ctx_bf   = (__bf16*)(ws + ws_alloc(off, (size_t)B * L * C * 2));
  __bf16* Win_bf   = (__bf16*)(ws + ws_alloc(off, (size_t)TC * C * 2));
  __bf16* Wout_bf  = (__bf16*)(ws + ws_alloc(off, (size_t)C * C * 2));
  __bf16* Winit_bf = (__bf16*)(ws + ws_alloc(off, (size_t)C * C * 2));
  __bf16* Wih_bf   = (__bf16*)(ws + ws_alloc(off, (size_t)G3 * XD * 2));
  __bf16* Whh_bf   = (__bf16*)(ws + ws_alloc(off, (size_t)G3 * C * 2));
  __bf16* qkv_bf   = (__bf16*)(ws + ws_alloc(off, (size_t)B * L * TC * 2));
  __bf16* vT_bf    = (__bf16*)(ws + ws_alloc(off, (size_t)B * C * L * 2));
  float*  rowmax   = (float*) (ws + ws_alloc(off, (size_t)B * H * L * 4));
  float*  rowsum   = (float*) (ws + ws_alloc(off, (size_t)B * H * L * 4));
  float*  colsum   = (float*) (ws + ws_alloc(off, (size_t)B * L * 4));
  __bf16* attnO    = (__bf16*)(ws + ws_alloc(off, (size_t)B * L * C * 2));
  float*  attnC    = (float*) (ws + ws_alloc(off, (size_t)B * L * C * 4));
  __bf16* summ_bf  = (__bf16*)(ws + ws_alloc(off, (size_t)B * C * 2));
  __bf16* xbuf     = (__bf16*)(ws + ws_alloc(off, (size_t)B * XD * 2));
  float*  hbuf     = (float*) (ws + ws_alloc(off, (size_t)B * C * 4));
  __bf16* hbf      = (__bf16*)(ws + ws_alloc(off, (size_t)B * C * 2));
  float*  gi       = (float*) (ws + ws_alloc(off, (size_t)B * G3 * 4));
  float*  gh       = (float*) (ws + ws_alloc(off, (size_t)B * G3 * 4));

  auto cvt = [&](const float* src, __bf16* dst, int n) {
    k_cvt_f32_bf16<<<(n + 255) / 256, 256, 0, stream>>>(src, dst, n);
  };

  // 1) fp32 -> bf16 staging (context + all GEMM weights)
  cvt(context, ctx_bf, B * L * C);
  cvt(W_in,   Win_bf,   TC * C);
  cvt(W_out,  Wout_bf,  C * C);
  cvt(W_init, Winit_bf, C * C);
  cvt(W_ih,   Wih_bf,   G3 * XD);
  cvt(W_hh,   Whh_bf,   G3 * C);

  // 2) QKV projection: [B*L, TC] = ctx @ W_in^T + b_in  (bf16 out)
  k_gemm_tdm<8, 0, 1><<<dim3(TC / 64, (B * L) / 128), 256, 0, stream>>>(
      ctx_bf, C, Win_bf, C, b_in, nullptr, qkv_bf, TC, C);

  // 3) V -> vT (K-contiguous layout for P@V); attention pass 1
  k_transpose_v<<<(B * L * C + 255) / 256, 256, 0, stream>>>(qkv_bf, vT_bf);
  k_attn_pass1<<<dim3(L / 16, H, B), 32, 0, stream>>>(qkv_bf, rowmax, rowsum);

  // 4) zero colsum, then pass 2 (O = P@V, colsum atomics)
  k_zero_f32<<<(B * L + 255) / 256, 256, 0, stream>>>(colsum, B * L);
  k_attn_pass2<<<dim3(L / 16, H, B), 32, 0, stream>>>(qkv_bf, vT_bf, rowmax,
                                                      rowsum, colsum, attnO);

  // 5) output projection: attnC = attnO @ W_out^T + b_out  (f32 out)
  k_gemm_tdm<8, 1, 0><<<dim3(C / 64, (B * L) / 128), 256, 0, stream>>>(
      attnO, C, Wout_bf, C, b_out, attnC, nullptr, C, C);

  // 6) key-softmax summary + inp0 + static concat
  k_summary<<<B, 256, 0, stream>>>(colsum, attnC, context, statice, summ_bf, xbuf);

  // 7) h0 = summary @ W_init^T + b_init  (f32 + bf16 out)
  k_gemm_tdm<4, 1, 1><<<dim3(C / 64, 1), 128, 0, stream>>>(
      summ_bf, C, Winit_bf, C, b_init, hbuf, hbf, C, C);

  // 8) 64 autoregressive GRU steps
  for (int t = 0; t < T; ++t) {
    k_gemm_tdm<4, 1, 0><<<dim3(G3 / 64, 1), 128, 0, stream>>>(
        xbuf, XD, Wih_bf, XD, b_ih, gi, nullptr, G3, XD);
    k_gemm_tdm<4, 1, 0><<<dim3(G3 / 64, 1), 128, 0, stream>>>(
        hbf, C, Whh_bf, C, b_hh, gh, nullptr, G3, C);
    k_gru_update<<<(B * C + 255) / 256, 256, 0, stream>>>(gi, gh, hbuf, hbf);
    k_head_step<<<B, 256, 0, stream>>>(hbuf, W_head, b_head, W_tf, b_tf,
                                       xbuf, out, t);
  }
}
